// CavAttention_5239860101892
// MI455X (gfx1250) — matrix operand order repaired and checked by reading, not compiled, not run on hardware
//
#include <hip/hip_runtime.h>

typedef __attribute__((ext_vector_type(16))) __bf16 v16bf;
typedef __attribute__((ext_vector_type(8)))  __bf16 v8bf;
typedef __attribute__((ext_vector_type(4)))  __bf16 v4bf;
typedef __attribute__((ext_vector_type(8)))  float  v8f;
typedef __attribute__((ext_vector_type(4)))  float  v4f;

namespace {
constexpr int kB = 2, kL = 5, kH = 48, kW = 176, kC = 256;
constexpr int kHeads = 8, kDimHead = 32;
constexpr int kInner = kHeads * kDimHead;     // 256
constexpr int kQkvN  = 3 * kInner;            // 768
constexpr int kHW    = kH * kW;               // 8448
constexpr int kNPix  = kB * kH * kW;          // 16896
constexpr int kPixPerWG = 16;
constexpr int kTok = kPixPerWG * kL;          // 80 tokens per WG
constexpr int kMT  = kTok / 16;               // 5 M tiles (80 = 5*16 exactly)
constexpr float kScale = 0.17677669529663687f; // 32^-0.5
}

// ---- CDNA5 async global<->LDS ops (ASYNCcnt-tracked) ----------------------
__device__ __forceinline__ void async_load_b128(unsigned lds_off, unsigned voff,
                                                const void* sbase) {
    asm volatile("global_load_async_to_lds_b128 %0, %1, %2"
                 :: "v"(lds_off), "v"(voff), "s"(sbase) : "memory");
}
__device__ __forceinline__ void async_store_b128(unsigned voff, unsigned lds_off,
                                                 void* sbase) {
    asm volatile("global_store_async_from_lds_b128 %0, %1, %2"
                 :: "v"(voff), "v"(lds_off), "s"(sbase) : "memory");
}
__device__ __forceinline__ void wait_async0() {
    asm volatile("s_wait_asynccnt 0x0" ::: "memory");
}

// Build a 16-element bf16 WMMA fragment from two contiguous 8-element chunks.
__device__ __forceinline__ v16bf make_frag(const __bf16* p0, const __bf16* p1) {
    v8bf lo = *(const v8bf*)p0;
    v8bf hi = *(const v8bf*)p1;
    v16bf r;
#pragma unroll
    for (int i = 0; i < 8; ++i) { r[i] = lo[i]; r[i + 8] = hi[i]; }
    return r;
}

// Element offset of token (gp, l) in the (B,L,H,W,C) tensor.
__device__ __forceinline__ int tok_elem_off(int gp, int l) {
    int b  = gp / kHW;
    int hw = gp - b * kHW;
    return ((b * kL + l) * kHW + hw) * kC;
}

// ---------------------------------------------------------------------------
// Prep: transpose + downconvert weights to bf16 in workspace.
//   wqkvT: [768][256] bf16 ;  woutT: [256][256] bf16
// ---------------------------------------------------------------------------
__global__ void prep_weights(const float* __restrict__ wqkv,
                             const float* __restrict__ wout,
                             __bf16* __restrict__ wqkvT,
                             __bf16* __restrict__ woutT) {
    int idx = blockIdx.x * blockDim.x + threadIdx.x;
    if (idx < kQkvN * kC) {
        int n = idx >> 8, k = idx & 255;
        wqkvT[idx] = (__bf16)wqkv[k * kQkvN + n];
    }
    int idx2 = idx - kQkvN * kC;
    if (idx2 >= 0 && idx2 < kC * kC) {
        int n = idx2 >> 8, k = idx2 & 255;
        woutT[idx2] = (__bf16)wout[k * kC + n];
    }
}

// ---------------------------------------------------------------------------
// Fused: x -> (QKV gemm) -> masked softmax attention -> (out gemm + bias)
// One workgroup = 16 pixels = 80 tokens. 256 threads = 8 waves.
// LDS map (160 KB):
//   [0 .. 40KB)    xs  [80][256] bf16   (A tile; reused as attention output o)
//   [40 .. 160KB)  multi-use region:
//        phase 0: f32 x staging [80][256] (async DMA target, 80KB)
//        phase 1-2: qkv [80][768] bf16 (120KB)
//        phase 3: f32 result tile [80][256] (async DMA source, 80KB)
// ---------------------------------------------------------------------------
__global__ void __launch_bounds__(256, 1)
cav_attention_fused(const float* __restrict__ x,
                    const int*   __restrict__ mask,
                    const __bf16* __restrict__ wqkvT,
                    const __bf16* __restrict__ woutT,
                    const float* __restrict__ bout,
                    float* __restrict__ out) {
    extern __shared__ char smem[];
    __bf16* xs   = (__bf16*)smem;                      // [80][256] bf16
    char*   big  = smem + kTok * kC * 2;               // 120KB multi-use
    __bf16* qkv  = (__bf16*)big;                       // [80][768] bf16

    const int tid  = threadIdx.x;
    const int wave = tid >> 5;
    const int lane = tid & 31;
    const int lane16 = lane & 15;
    const bool lo_half = lane < 16;
    const int pix0 = blockIdx.x * kPixPerWG;

    // ---- Phase 0a: async DMA x tokens (f32) into LDS staging --------------
    // 80 tokens * 64 16B-chunks = 5120 chunks; 20 per thread.
    {
        unsigned xf_base = (unsigned)(size_t)(void*)big;
#pragma unroll 4
        for (int i = 0; i < 20; ++i) {
            int chunk = i * 256 + tid;
            int t   = chunk >> 6;
            int c16 = chunk & 63;
            int p = t / kL, l = t - p * kL;
            unsigned goff = (unsigned)(tok_elem_off(pix0 + p, l) * 4 + c16 * 16);
            unsigned loff = xf_base + (unsigned)(t * kC + c16 * 4) * 4;
            async_load_b128(loff, goff, x);
        }
        wait_async0();
    }
    __syncthreads();

    // ---- Phase 0b: LDS f32 -> LDS bf16 conversion (vectorized) ------------
    {
        const v4f* xfv = (const v4f*)big;
        v4bf* xsv = (v4bf*)xs;
#pragma unroll 4
        for (int i = 0; i < 20; ++i) {
            int idx = i * 256 + tid;
            v4f f = xfv[idx];
            v4bf h;
#pragma unroll
            for (int e = 0; e < 4; ++e) h[e] = (__bf16)f[e];
            xsv[idx] = h;
        }
    }
    __syncthreads();

    // ---- Phase 1: qkv[80][768] = xs[80][256] @ wqkv  (bf16 WMMA) ----------
    // Wave w owns N-tiles {w, w+8, ..., w+40}. A fragments cached per M-tile.
    for (int mt = 0; mt < kMT; ++mt) {
        const __bf16* arow = xs + (mt * 16 + lane16) * kC;
        const int koff = lo_half ? 0 : 8;
        v16bf afrag[8];
#pragma unroll
        for (int ks = 0; ks < 8; ++ks)
            afrag[ks] = make_frag(arow + ks * 32 + koff,
                                  arow + ks * 32 + koff + 16);

        for (int ntj = 0; ntj < 6; ++ntj) {
            int nt = wave + ntj * 8;
            int n  = nt * 16 + lane16;
            const __bf16* brow = wqkvT + n * kC + (lo_half ? 0 : 8);
            if (ntj + 1 < 6)
                __builtin_prefetch(wqkvT + (nt + 8) * 16 * kC, 0, 1);
            v8f acc = {};
#pragma unroll
            for (int ks = 0; ks < 8; ++ks) {
                v16bf bfrag = make_frag(brow + ks * 32, brow + ks * 32 + 16);
                acc = __builtin_amdgcn_wmma_f32_16x16x32_bf16(
                    false, afrag[ks], false, bfrag, (short)0, acc, false, false);
            }
            // scatter accumulator to qkv LDS (D layout: VGPR r -> row r / r+8)
            int rbase = mt * 16 + (lo_half ? 0 : 8);
            int col   = nt * 16 + lane16;
#pragma unroll
            for (int r = 0; r < 8; ++r)
                qkv[(rbase + r) * kQkvN + col] = (__bf16)acc[r];
        }
    }
    __syncthreads();

    // ---- Phase 2: per-(pixel, head) 5x5 masked softmax attention ----------
    if (tid < kPixPerWG * kHeads) {
        int p  = tid >> 3;
        int hd = tid & 7;
        int gp = pix0 + p;
        int tbase = p * kL;
        int mk[kL];
#pragma unroll
        for (int j = 0; j < kL; ++j) mk[j] = mask[gp * kL + j];

        float att[kL][kL];
#pragma unroll
        for (int i = 0; i < kL; ++i) {
            const __bf16* qi = qkv + (tbase + i) * kQkvN + hd * kDimHead;
#pragma unroll
            for (int j = 0; j < kL; ++j) {
                const __bf16* kj = qkv + (tbase + j) * kQkvN + kInner + hd * kDimHead;
                float s = 0.f;
#pragma unroll
                for (int d = 0; d < kDimHead; ++d)
                    s += (float)qi[d] * (float)kj[d];
                att[i][j] = mk[j] ? s * kScale : -3.0e38f;
            }
            float mmax = att[i][0];
#pragma unroll
            for (int j = 1; j < kL; ++j) mmax = fmaxf(mmax, att[i][j]);
            float sum = 0.f;
#pragma unroll
            for (int j = 0; j < kL; ++j) {
                float e = __expf(att[i][j] - mmax);
                att[i][j] = e;
                sum += e;
            }
            float inv = 1.0f / sum;
#pragma unroll
            for (int j = 0; j < kL; ++j) att[i][j] *= inv;
        }
        // o = att @ v  -> write bf16 into the retired xs region
#pragma unroll
        for (int i = 0; i < kL; ++i) {
#pragma unroll 8
            for (int d = 0; d < kDimHead; ++d) {
                float o = 0.f;
#pragma unroll
                for (int j = 0; j < kL; ++j)
                    o += att[i][j] *
                         (float)qkv[(tbase + j) * kQkvN + 2 * kInner + hd * kDimHead + d];
                xs[(tbase + i) * kC + hd * kDimHead + d] = (__bf16)o;
            }
        }
    }
    __syncthreads();

    // ---- Phase 3: res = o[80][256] @ w_out + b_out  (bf16 WMMA) -----------
    // Wave w owns N-tiles {2w, 2w+1}. Results go to an LDS f32 tile.
    float* res = (float*)big;   // qkv region is dead now
    for (int mt = 0; mt < kMT; ++mt) {
        const __bf16* arow = xs + (mt * 16 + lane16) * kC;
        const int koff = lo_half ? 0 : 8;
        v16bf afrag[8];
#pragma unroll
        for (int ks = 0; ks < 8; ++ks)
            afrag[ks] = make_frag(arow + ks * 32 + koff,
                                  arow + ks * 32 + koff + 16);

        for (int nti = 0; nti < 2; ++nti) {
            int nt = wave * 2 + nti;
            int n  = nt * 16 + lane16;
            const __bf16* brow = woutT + n * kC + (lo_half ? 0 : 8);
            v8f acc = {};
#pragma unroll
            for (int ks = 0; ks < 8; ++ks) {
                v16bf bfrag = make_frag(brow + ks * 32, brow + ks * 32 + 16);
                acc = __builtin_amdgcn_wmma_f32_16x16x32_bf16(
                    false, afrag[ks], false, bfrag, (short)0, acc, false, false);
            }
            float bias = bout[n];
            int rbase = mt * 16 + (lo_half ? 0 : 8);
#pragma unroll
            for (int r = 0; r < 8; ++r)
                res[(rbase + r) * kC + n] = acc[r] + bias;
        }
    }
    __syncthreads();

    // ---- Phase 4: async DMA result tile LDS -> global (coalesced) ---------
    {
        unsigned res_base = (unsigned)(size_t)(void*)res;
#pragma unroll 4
        for (int i = 0; i < 20; ++i) {
            int chunk = i * 256 + tid;
            int t   = chunk >> 6;
            int c16 = chunk & 63;
            int p = t / kL, l = t - p * kL;
            unsigned goff = (unsigned)(tok_elem_off(pix0 + p, l) * 4 + c16 * 16);
            unsigned loff = res_base + (unsigned)(t * kC + c16 * 4) * 4;
            async_store_b128(goff, loff, out);
        }
        wait_async0();
    }
}

// ---------------------------------------------------------------------------
extern "C" void kernel_launch(void* const* d_in, const int* in_sizes, int n_in,
                              void* d_out, int out_size, void* d_ws, size_t ws_size,
                              hipStream_t stream) {
    const float* x    = (const float*)d_in[0];
    const int*   mask = (const int*)d_in[1];
    const float* wqkv = (const float*)d_in[2];
    const float* wout = (const float*)d_in[3];
    const float* bout = (const float*)d_in[4];
    float* out = (float*)d_out;

    __bf16* wqkvT = (__bf16*)d_ws;
    __bf16* woutT = (__bf16*)((char*)d_ws + (size_t)kQkvN * kC * 2);

    int prepN = kQkvN * kC + kC * kC;   // 262144
    prep_weights<<<dim3((prepN + 255) / 256), dim3(256), 0, stream>>>(
        wqkv, wout, wqkvT, woutT);

    size_t lds_bytes = (size_t)kTok * kC * 2 + (size_t)kTok * kQkvN * 2;  // 160 KB
    cav_attention_fused<<<dim3(kNPix / kPixPerWG), dim3(256), lds_bytes, stream>>>(
        x, mask, wqkvT, woutT, bout, out);
}
